// CCAttention_2207613190641
// MI455X (gfx1250) — matrix operand order repaired and checked by reading, not compiled, not run on hardware
//
#include <hip/hip_runtime.h>
#include <hip/hip_bf16.h>

// ---- problem constants ----
#define B_   8
#define C_   288
#define H_   128
#define W_   128
#define N_   (H_ * W_)     // 16384
#define CQ_  36
#define CQP_ 64            // q/k channels padded for WMMA K-depth
#define RP_  416           // packed weight rows: 64(q) + 64(k) + 288(v)
#define KP_  296           // qkv LDS k-stride in ushorts (592 B, 16B-aligned)
#define QS_  80            // logits LDS row stride in ushorts (64 + 32B pad)
#define VS_  136           // outgemm LDS row stride in ushorts (128 + 16B pad)

// ---- CDNA5 WMMA / TDM types ----
typedef __attribute__((ext_vector_type(16))) __bf16          v16bf;
typedef __attribute__((ext_vector_type(16))) unsigned short  v16u;
typedef __attribute__((ext_vector_type(8)))  unsigned short  u16x8;
typedef __attribute__((ext_vector_type(8)))  float           v8f;
typedef __attribute__((ext_vector_type(4)))  unsigned int    u32x4;
typedef __attribute__((ext_vector_type(8)))  int             i32x8;
typedef __attribute__((ext_vector_type(4)))  int             i32x4;

__device__ __forceinline__ unsigned short f2bf(float f) {
    unsigned u = __float_as_uint(f);
    return (unsigned short)((u + 0x7FFFu + ((u >> 16) & 1u)) >> 16);
}

// A (16x32 bf16): lane-half kh8 in {0,8}; chunks at K = kh8+[0..7], kh8+16+[0..7]
// B (32x16 bf16): lane-half kh8 in {0,8}; chunks at K = kh8+[0..7], kh8+16+[0..7]
__device__ __forceinline__ v16u pack2(u16x8 c0, u16x8 c1) {
    v16u f;
    #pragma unroll
    for (int e = 0; e < 8; ++e) { f[e] = c0[e]; f[e + 8] = c1[e]; }
    return f;
}

__device__ __forceinline__ v8f wmma_bf16(v16u a, v16u b, v8f c) {
    return __builtin_amdgcn_wmma_f32_16x16x32_bf16(
        false, __builtin_bit_cast(v16bf, a),
        false, __builtin_bit_cast(v16bf, b),
        (short)0, c, false, false);
}

// ------------------------------------------------------------------
// TDM: 2D tile load global -> LDS (D# per CDNA5 ISA ch.8).
// data_size = 2B. LDS row padding codes per ISA:
//   pad_interval code c -> pad every 2^(c+1) DWORDs; pad_amount code a -> a+1 DWORDs
// ------------------------------------------------------------------
__device__ __forceinline__ void tdm_load_2d(
    unsigned lds_off, const void* gptr,
    unsigned tile_d0, unsigned tile_d1,
    unsigned tensor_d0, unsigned tensor_d1,
    unsigned long long stride0_elems,
    unsigned pad_interval_code, unsigned pad_amount_code, unsigned pad_en) {
    unsigned long long ga = (unsigned long long)(size_t)gptr;
    u32x4 g0;
    g0[0] = 1u;                                    // count=1, user mode
    g0[1] = lds_off;                               // LDS byte address
    g0[2] = (unsigned)(ga & 0xFFFFFFFFu);
    g0[3] = (unsigned)((ga >> 32) & 0x01FFFFFFu) | 0x80000000u;  // type=2
    i32x8 g1;
    g1[0] = (int)((1u << 16) |                     // data_size = 2B
                  (pad_en << 20) |
                  (pad_interval_code << 22) |
                  (pad_amount_code << 25));
    g1[1] = (int)((tensor_d0 & 0xFFFFu) << 16);                    // dim0 lo16
    g1[2] = (int)(((tensor_d0 >> 16) & 0xFFFFu) |
                  ((tensor_d1 & 0xFFFFu) << 16));                  // dim0 hi / dim1 lo
    g1[3] = (int)(((tensor_d1 >> 16) & 0xFFFFu) | (tile_d0 << 16));
    g1[4] = (int)(tile_d1 & 0xFFFFu);                              // tile_dim1, dim2=0
    g1[5] = (int)(stride0_elems & 0xFFFFFFFFu);
    g1[6] = (int)((stride0_elems >> 32) & 0xFFFFu);
    g1[7] = 0;
    i32x4 z4 = {0, 0, 0, 0};
#if __clang_major__ >= 23
    i32x8 z8 = {0, 0, 0, 0, 0, 0, 0, 0};
    __builtin_amdgcn_tensor_load_to_lds(g0, g1, z4, z4, z8, 0);
#else
    __builtin_amdgcn_tensor_load_to_lds(g0, g1, z4, z4, 0);
#endif
}

// ------------------------------------------------------------------
// K0: pack Wq/Wk/Wv -> zero-padded bf16 Wp[416][288], biases -> bp[416]
// ------------------------------------------------------------------
__global__ __launch_bounds__(256) void pack_kernel(
    const float* __restrict__ Wq, const float* __restrict__ bq,
    const float* __restrict__ Wk, const float* __restrict__ bk,
    const float* __restrict__ Wv, const float* __restrict__ bv,
    unsigned short* __restrict__ Wp, float* __restrict__ bp) {
    int i = blockIdx.x * 256 + threadIdx.x;
    if (i < RP_ * C_) {
        int row = i / C_, k = i % C_;
        float v;
        if (row < 64)       v = (row < CQ_) ? Wq[row * C_ + k] : 0.f;
        else if (row < 128) { int r = row - 64; v = (r < CQ_) ? Wk[r * C_ + k] : 0.f; }
        else                v = Wv[(row - 128) * C_ + k];
        Wp[i] = f2bf(v);
    }
    if (i < RP_) {
        float v;
        if (i < 64)         v = (i < CQ_) ? bq[i] : 0.f;
        else if (i < 128)   { int r = i - 64; v = (r < CQ_) ? bk[r] : 0.f; }
        else                v = bv[i - 128];
        bp[i] = v;
    }
}

// ------------------------------------------------------------------
// K1: y[b,c] = mean over H*W of x
// ------------------------------------------------------------------
__global__ __launch_bounds__(256) void mean_kernel(const float* __restrict__ x,
                                                   float* __restrict__ y) {
    __shared__ float red[256];
    const int bc = blockIdx.x;
    const float4* xp = (const float4*)(x + (size_t)bc * N_);
    float s = 0.f;
    for (int i = threadIdx.x; i < N_ / 4; i += 256) {
        float4 v = xp[i];
        s += v.x + v.y + v.z + v.w;
    }
    red[threadIdx.x] = s;
    __syncthreads();
    for (int o = 128; o > 0; o >>= 1) {
        if (threadIdx.x < o) red[threadIdx.x] += red[threadIdx.x + o];
        __syncthreads();
    }
    if (threadIdx.x == 0) y[bc] = red[0] * (1.0f / (float)N_);
}

// ------------------------------------------------------------------
// K2: ECA width-3 conv over channels + sigmoid
// ------------------------------------------------------------------
__global__ __launch_bounds__(256) void eca_kernel(const float* __restrict__ y,
                                                  const float* __restrict__ ew,
                                                  float* __restrict__ sig) {
    int i = blockIdx.x * 256 + threadIdx.x;
    if (i >= B_ * C_) return;
    int b = i / C_, c = i % C_;
    float l = (c > 0)      ? y[b * C_ + c - 1] : 0.f;
    float m = y[i];
    float r = (c < C_ - 1) ? y[b * C_ + c + 1] : 0.f;
    float v = ew[0] * l + ew[1] * m + ew[2] * r;
    sig[i] = 1.f / (1.f + __expf(-v));
}

// ------------------------------------------------------------------
// K3: fused q,k,v projection GEMM; x tile (288K x 64N) staged once into LDS
// (f32->bf16 conversion happens here, so manual staging not TDM).
// Wave index pinned to SGPR so the segment branches are scalar.
// ------------------------------------------------------------------
__global__ __launch_bounds__(256) void qkv_kernel(
    const float* __restrict__ x,
    const unsigned short* __restrict__ Wp, const float* __restrict__ bp,
    unsigned short* __restrict__ qT, unsigned short* __restrict__ kT,
    unsigned short* __restrict__ vbf, unsigned short* __restrict__ vbfT) {
    __shared__ unsigned short xs[64 * KP_];   // 37,888 B

    const int tid  = threadIdx.x;
    const int lane = tid & 31;
    const int wv   = __builtin_amdgcn_readfirstlane(tid >> 5);  // SGPR wave id
    const int b    = blockIdx.x >> 8;
    const int n0   = (blockIdx.x & 255) * 64;

    const float* xb = x + (size_t)b * C_ * N_ + n0;
    for (int i = tid; i < C_ * 16; i += 256) {
        int k  = i >> 4;
        int c4 = (i & 15) * 4;
        float4 v = *(const float4*)(xb + (size_t)k * N_ + c4);
        xs[(c4 + 0) * KP_ + k] = f2bf(v.x);
        xs[(c4 + 1) * KP_ + k] = f2bf(v.y);
        xs[(c4 + 2) * KP_ + k] = f2bf(v.z);
        xs[(c4 + 3) * KP_ + k] = f2bf(v.w);
    }
    __syncthreads();

    const int mrow = lane & 15, ncol = lane & 15;
    const int kh8  = (lane >> 4) << 3;
    const int kh16 = (lane >> 4) << 4;
    const int rofs = (lane >> 4) << 3;

    for (int t = wv; t < 104; t += 8) {
        const int mt = t >> 2, ns = t & 3;   // scalar (t derived from SGPR wv)
        const unsigned short* ap  = Wp + (size_t)(mt * 16 + mrow) * C_;
        const unsigned short* bq2 = xs + (size_t)(ns * 16 + ncol) * KP_;

        v8f acc = {};
        #pragma unroll
        for (int k0 = 0; k0 < C_; k0 += 32) {
            u16x8 a0 = *(const u16x8*)(ap + k0 + kh8);
            u16x8 a1 = *(const u16x8*)(ap + k0 + kh8 + 16);
            u16x8 b0 = *(const u16x8*)(bq2 + k0 + kh8);
            u16x8 b1 = *(const u16x8*)(bq2 + k0 + kh8 + 16);
            acc = wmma_bf16(pack2(a0, a1), pack2(b0, b1), acc);
        }

        const int n_g = n0 + ns * 16 + ncol;   // spatial index h*W + w
        const int hh  = n_g >> 7, ww = n_g & 127;
        if (mt < 4) {                            // q rows (scalar branch)
            const size_t ob = ((size_t)b * N_ + n_g) * CQP_;
            #pragma unroll
            for (int r = 0; r < 8; ++r) {
                int row = mt * 16 + r + rofs;
                qT[ob + row] = f2bf(acc[r] + bp[row]);
            }
        } else if (mt < 8) {                     // k rows
            const size_t ob = ((size_t)b * N_ + n_g) * CQP_;
            #pragma unroll
            for (int r = 0; r < 8; ++r) {
                int row = mt * 16 + r + rofs;
                kT[ob + (row - 64)] = f2bf(acc[r] + bp[row]);
            }
        } else {                                 // v rows
            #pragma unroll
            for (int r = 0; r < 8; ++r) {
                int row = mt * 16 + r + rofs;
                int c   = row - 128;
                unsigned short val = f2bf(acc[r] + bp[row]);
                vbf [(size_t)(b * C_ + c) * N_ + n_g]          = val;
                vbfT[(size_t)(b * C_ + c) * N_ + ww * H_ + hh] = val;
            }
        }
    }
}

// ------------------------------------------------------------------
// K4: attention logits E = q.k (K=64); one block per (b,pos) slab.
// q & k slabs (128 rows x 64 ch) staged into LDS by the Tensor Data Mover
// (strided-row gather for alongH), padded to 80-ushort rows. Each wave owns
// one mt (SGPR), hoists A fragments, loops 8 nt tiles.
// ------------------------------------------------------------------
__global__ __launch_bounds__(256) void logits_kernel(
    const unsigned short* __restrict__ qT,
    const unsigned short* __restrict__ kT,
    float* __restrict__ eOut, int alongH) {
    __shared__ unsigned short qs[128 * QS_];   // 20,480 B
    __shared__ unsigned short ks[128 * QS_];   // 20,480 B

    const int lane = threadIdx.x & 31;
    const int wv   = __builtin_amdgcn_readfirstlane(threadIdx.x >> 5);
    const int b    = blockIdx.x >> 7;
    const int pos  = blockIdx.x & 127;

    // TDM staging: rows m=0..127 are spatial sm = alongH ? m*W+pos : pos*W+m
    if (wv == 0) {
        const size_t start = alongH ? (size_t)pos * CQP_
                                    : (size_t)pos * W_ * CQP_;
        const unsigned long long stride0 = alongH ? (unsigned long long)W_ * CQP_
                                                  : (unsigned long long)CQP_;
        const size_t base = (size_t)b * N_ * CQP_ + start;
        // rows: 64 elems (128B=32 DW) + 32B pad (8 DW) -> codes 4 / 7
        tdm_load_2d((unsigned)(size_t)&qs[0], qT + base, 64, 128,
                    64, 1u << 20, stride0, 4, 7, 1);
        tdm_load_2d((unsigned)(size_t)&ks[0], kT + base, 64, 128,
                    64, 1u << 20, stride0, 4, 7, 1);
        __builtin_amdgcn_s_wait_tensorcnt(0);
    }
    __syncthreads();

    const int mrow = lane & 15, ncol = lane & 15;
    const int kh8  = (lane >> 4) << 3;
    const int kh16 = (lane >> 4) << 4;
    const int rofs = (lane >> 4) << 3;
    const int mt   = wv;

    // hoist A fragments (2 K-steps)
    v16u afrag[2];
    #pragma unroll
    for (int k = 0; k < 2; ++k) {
        const unsigned short* ap = qs + (mt * 16 + mrow) * QS_ + k * 32;
        afrag[k] = pack2(*(const u16x8*)(ap + kh8),
                         *(const u16x8*)(ap + kh8 + 16));
    }

    const size_t obase = (size_t)(b * 128 + pos) * 128 * 128;
    for (int nt = 0; nt < 8; ++nt) {
        const unsigned short* bp2 = ks + (nt * 16 + ncol) * QS_;
        v8f acc = {};
        #pragma unroll
        for (int k = 0; k < 2; ++k) {
            v16u bf = pack2(*(const u16x8*)(bp2 + k * 32 + kh8),
                            *(const u16x8*)(bp2 + k * 32 + kh8 + 16));
            acc = wmma_bf16(afrag[k], bf, acc);
        }
        const int ng = nt * 16 + ncol;
        #pragma unroll
        for (int r = 0; r < 8; ++r) {
            int m = mt * 16 + r + rofs;
            float val = acc[r];
            if (alongH && m == ng) val = -1e30f;
            eOut[obase + (size_t)m * 128 + ng] = val;
        }
    }
}

// ------------------------------------------------------------------
// K5: joint softmax over [eH row | eW row] per (b,h,w); one wave per row.
// Coalesced reads and writes; att rows stored contiguously.
// ------------------------------------------------------------------
__global__ __launch_bounds__(256) void softmax_kernel(
    const float* __restrict__ eH, const float* __restrict__ eW,
    unsigned short* __restrict__ attH, unsigned short* __restrict__ attW) {
    const int lane = threadIdx.x & 31;
    const int wv   = __builtin_amdgcn_readfirstlane(threadIdx.x >> 5);
    const int row  = blockIdx.x * 8 + wv;     // scalar row id
    const int b    = row >> 14;
    const int rem  = row & 16383;
    const int h    = rem >> 7, w = rem & 127;

    const size_t hbase = ((size_t)(b * 128 + w) * 128 + h) * 128;
    const size_t wbase = ((size_t)(b * 128 + h) * 128 + w) * 128;

    float vh[4], vw[4];
    float mx = -3.4e38f;
    #pragma unroll
    for (int i = 0; i < 4; ++i) {
        vh[i] = eH[hbase + i * 32 + lane];
        vw[i] = eW[wbase + i * 32 + lane];
        mx = fmaxf(mx, fmaxf(vh[i], vw[i]));
    }
    #pragma unroll
    for (int o = 16; o > 0; o >>= 1) mx = fmaxf(mx, __shfl_xor(mx, o, 32));
    float s = 0.f;
    #pragma unroll
    for (int i = 0; i < 4; ++i) {
        vh[i] = __expf(vh[i] - mx);
        vw[i] = __expf(vw[i] - mx);
        s += vh[i] + vw[i];
    }
    #pragma unroll
    for (int o = 16; o > 0; o >>= 1) s += __shfl_xor(s, o, 32);
    const float inv = 1.f / s;

    #pragma unroll
    for (int i = 0; i < 4; ++i) {
        int j = i * 32 + lane;
        attH[hbase + j] = f2bf(vh[i] * inv);
        attW[wbase + j] = f2bf(vw[i] * inv);
    }
}

// ------------------------------------------------------------------
// K6: out = V x att^T (K=128); one block per (b,pos) slab.
// v slab (288ch x 128) TDM-staged into LDS in two 144-channel halves
// (strided rows, stride N); each wave owns one st (SGPR), hoists its 4
// B-fragments, loops 9 c-tiles per half -> 72 WMMA/wave.
// alongH writes accH; alongW fuses final combine.
// ------------------------------------------------------------------
__global__ __launch_bounds__(256) void outgemm_kernel(
    const unsigned short* __restrict__ vsel,   // vbfT (alongH) or vbf (alongW)
    const unsigned short* __restrict__ att,    // attH or attW (row-contig)
    float* __restrict__ accH,
    const float* __restrict__ x, const float* __restrict__ sig,
    const float* __restrict__ gamma, float* __restrict__ out,
    int alongH) {
    __shared__ unsigned short vs[144 * VS_];   // 39,168 B

    const int lane = threadIdx.x & 31;
    const int wv   = __builtin_amdgcn_readfirstlane(threadIdx.x >> 5);
    const int b    = blockIdx.x >> 7;
    const int pos  = blockIdx.x & 127;
    const int st   = wv;

    const int mrow = lane & 15, ncol = lane & 15;
    const int kh8  = (lane >> 4) << 3;
    const int kh16 = (lane >> 4) << 4;
    const int rofs = (lane >> 4) << 3;

    // hoist B fragments (4 K-steps) from att rows (each row read once/block)
    const unsigned short* bp2 = att + ((size_t)(b * 128 + pos) * 128 +
                                       (st * 16 + ncol)) * 128;
    v16u bfrag[4];
    #pragma unroll
    for (int k = 0; k < 4; ++k)
        bfrag[k] = pack2(*(const u16x8*)(bp2 + k * 32 + kh8),
                        *(const u16x8*)(bp2 + k * 32 + kh8 + 16));

    const float g = gamma[0];

    for (int hf = 0; hf < 2; ++hf) {
        __syncthreads();   // previous half fully consumed
        if (wv == 0) {
            // tile: 144 rows x 128 elems, row stride N_ elems in global;
            // rows: 128 elems (256B=64 DW) + 16B pad (4 DW) -> codes 5 / 3
            const unsigned short* src = vsel +
                (size_t)(b * C_ + hf * 144) * N_ + (size_t)pos * 128;
            tdm_load_2d((unsigned)(size_t)&vs[0], src, 128, 144,
                        128, 1u << 20, (unsigned long long)N_, 5, 3, 1);
            __builtin_amdgcn_s_wait_tensorcnt(0);
        }
        __syncthreads();

        for (int ct = 0; ct < 9; ++ct) {
            const unsigned short* ap = vs + (ct * 16 + mrow) * VS_;
            v8f acc = {};
            #pragma unroll
            for (int k = 0; k < 4; ++k) {
                v16u af = pack2(*(const u16x8*)(ap + k * 32 + kh8),
                                *(const u16x8*)(ap + k * 32 + kh8 + 16));
                acc = wmma_bf16(af, bfrag[k], acc);
            }

            const int cbase = hf * 144 + ct * 16;
            if (alongH) {
                #pragma unroll
                for (int r = 0; r < 8; ++r) {
                    int c  = cbase + r + rofs;
                    int hh = st * 16 + ncol;
                    accH[(size_t)(b * C_ + c) * N_ + (size_t)hh * W_ + pos] = acc[r];
                }
            } else {
                #pragma unroll
                for (int r = 0; r < 8; ++r) {
                    int c  = cbase + r + rofs;
                    int ww = st * 16 + ncol;
                    size_t idx = (size_t)(b * C_ + c) * N_ + (size_t)pos * W_ + ww;
                    float xv = x[idx];
                    out[idx] = g * (accH[idx] + acc[r] + xv * sig[b * C_ + c]) + xv;
                }
            }
        }
    }
}

// ------------------------------------------------------------------
extern "C" void kernel_launch(void* const* d_in, const int* in_sizes, int n_in,
                              void* d_out, int out_size, void* d_ws, size_t ws_size,
                              hipStream_t stream) {
    const float* x    = (const float*)d_in[0];
    const float* Wq   = (const float*)d_in[1];
    const float* bq   = (const float*)d_in[2];
    const float* Wk   = (const float*)d_in[3];
    const float* bk   = (const float*)d_in[4];
    const float* Wv   = (const float*)d_in[5];
    const float* bv   = (const float*)d_in[6];
    const float* ecaw = (const float*)d_in[7];
    const float* gmm  = (const float*)d_in[8];
    float* out = (float*)d_out;

    const size_t SZ_QT  = (size_t)B_ * N_ * CQP_ * 2;
    const size_t SZ_VBF = (size_t)B_ * C_ * N_ * 2;
    const size_t SZ_E   = (size_t)B_ * 128 * 128 * 128 * 4;
    const size_t SZ_ATT = (size_t)B_ * 128 * 128 * 128 * 2;
    const size_t SZ_ACC = (size_t)B_ * C_ * N_ * 4;

    char* p = (char*)d_ws;
    unsigned short* qT   = (unsigned short*)p; p += SZ_QT;
    unsigned short* kT   = (unsigned short*)p; p += SZ_QT;
    unsigned short* vbf  = (unsigned short*)p; p += SZ_VBF;
    unsigned short* vbfT = (unsigned short*)p; p += SZ_VBF;
    float*          eH   = (float*)p;          p += SZ_E;
    float*          eW   = (float*)p;          p += SZ_E;
    unsigned short* attH = (unsigned short*)p; p += SZ_ATT;
    unsigned short* attW = (unsigned short*)p; p += SZ_ATT;
    float*          accH = (float*)p;          p += SZ_ACC;
    unsigned short* Wp   = (unsigned short*)p; p += (size_t)RP_ * C_ * 2;
    float*          bp   = (float*)p;          p += (size_t)RP_ * 4;
    float*          y    = (float*)p;          p += (size_t)B_ * C_ * 4;
    float*          sig  = (float*)p;          p += (size_t)B_ * C_ * 4;

    pack_kernel<<<(RP_ * C_ + 255) / 256, 256, 0, stream>>>(
        Wq, bq, Wk, bk, Wv, bv, Wp, bp);
    mean_kernel<<<B_ * C_, 256, 0, stream>>>(x, y);
    eca_kernel<<<(B_ * C_ + 255) / 256, 256, 0, stream>>>(y, ecaw, sig);

    qkv_kernel<<<B_ * 256, 256, 0, stream>>>(x, Wp, bp, qT, kT, vbf, vbfT);

    logits_kernel<<<B_ * 128, 256, 0, stream>>>(qT, kT, eH, 1);
    logits_kernel<<<B_ * 128, 256, 0, stream>>>(qT, kT, eW, 0);

    softmax_kernel<<<131072 / 8, 256, 0, stream>>>(eH, eW, attH, attW);

    outgemm_kernel<<<B_ * 128, 256, 0, stream>>>(
        vbfT, attH, accH, x, sig, gmm, out, 1);
    outgemm_kernel<<<B_ * 128, 256, 0, stream>>>(
        vbf, attW, accH, x, sig, gmm, out, 0);
}
